// NystromAttentionLayer_75780402970798
// MI455X (gfx1250) — compile-verified, matrix operand order checked by reading, not run on hardware
//
#include <hip/hip_runtime.h>

// ---------------------------------------------------------------------------
// Nystrom attention for MI455X (gfx1250), wave32, WMMA-based.
// All WMMA operands kept in [outer][K] LDS layouts so every fragment load is
// two contiguous ds_load_b128 per lane.
// ---------------------------------------------------------------------------

typedef __bf16 bf16;
typedef __bf16 v16bf __attribute__((ext_vector_type(16)));
typedef __bf16 v8bf  __attribute__((ext_vector_type(8)));
typedef float  v8f   __attribute__((ext_vector_type(8)));
typedef float  v2f   __attribute__((ext_vector_type(2)));

static constexpr int Bc = 4;       // batch
static constexpr int Hc = 16;      // heads
static constexpr int Lc = 4096;    // sequence
static constexpr int Dc = 64;      // head dim
static constexpr int DM = 1024;    // d_model
static constexpr int Mc = 128;     // landmarks
static constexpr int BHc = Bc * Hc;
static constexpr int Rc = Bc * Lc; // 16384 rows

// --- helpers ---------------------------------------------------------------

__device__ __forceinline__ void zero_v8(v8f& a) {
#pragma unroll
  for (int i = 0; i < 8; ++i) a[i] = 0.0f;
}

// 16-bit WMMA fragment (A, or B when LDS holds op(B) as [N][K]):
// per-lane data = 8 contiguous bf16 at k0+(lane>>4)*8 and 8 more at +16.
// p must be 16B aligned (strides used are multiples of 8 elements).
__device__ __forceinline__ v16bf lds_frag16(const bf16* p) {
  v8bf lo = *(const v8bf*)p;
  v8bf hi = *(const v8bf*)(p + 16);
  return __builtin_shufflevector(lo, hi, 0, 1, 2, 3, 4, 5, 6, 7,
                                 8, 9, 10, 11, 12, 13, 14, 15);
}

__device__ __forceinline__ const bf16* frag_ptr(const bf16* base, int stride,
                                                int row0, int k0, int lane) {
  return base + (size_t)(row0 + (lane & 15)) * stride + k0 + ((lane >> 4) << 3);
}

// Build a bf16 fragment directly from f32 global memory (row-major, row ptr p
// already offset to k0; needs p 16B aligned).
__device__ __forceinline__ v16bf global_frag_f32(const float* p) {
  float4 a0 = *(const float4*)(p);
  float4 a1 = *(const float4*)(p + 4);
  float4 b0 = *(const float4*)(p + 16);
  float4 b1 = *(const float4*)(p + 20);
  v16bf f;
  f[0] = (bf16)a0.x;  f[1] = (bf16)a0.y;  f[2] = (bf16)a0.z;  f[3] = (bf16)a0.w;
  f[4] = (bf16)a1.x;  f[5] = (bf16)a1.y;  f[6] = (bf16)a1.z;  f[7] = (bf16)a1.w;
  f[8] = (bf16)b0.x;  f[9] = (bf16)b0.y;  f[10] = (bf16)b0.z; f[11] = (bf16)b0.w;
  f[12] = (bf16)b1.x; f[13] = (bf16)b1.y; f[14] = (bf16)b1.z; f[15] = (bf16)b1.w;
  return f;
}

// convert float4 -> 4 bf16, store as 8 bytes
__device__ __forceinline__ void cvt_store4(bf16* dst, float4 v) {
  union { bf16 b[4]; uint2 u; } t;
  t.b[0] = (bf16)v.x; t.b[1] = (bf16)v.y; t.b[2] = (bf16)v.z; t.b[3] = (bf16)v.w;
  *(uint2*)dst = t.u;
}

// sum of the two bf16 halves of a dword, as f32
__device__ __forceinline__ float bf16pair_sum(unsigned u) {
  return __uint_as_float(u << 16) + __uint_as_float(u & 0xffff0000u);
}

// ---------------------------------------------------------------------------
// 0) Weight transpose + bf16 convert: Wt[n][k] = (bf16)W[k][n]   (1024x1024)
// ---------------------------------------------------------------------------
__global__ __launch_bounds__(256) void transpose_w(const float* __restrict__ W,
                                                   bf16* __restrict__ Wt) {
  __shared__ float t[32][33];
  int bx = blockIdx.x & 31, by = blockIdx.x >> 5;
  int lx = threadIdx.x & 31, ly = threadIdx.x >> 5;   // 32 x 8
#pragma unroll
  for (int j = 0; j < 32; j += 8)
    t[ly + j][lx] = W[(size_t)(by * 32 + ly + j) * DM + bx * 32 + lx];
  __syncthreads();
#pragma unroll
  for (int j = 0; j < 32; j += 8)
    Wt[(size_t)(bx * 32 + ly + j) * DM + by * 32 + lx] = (bf16)t[lx][ly + j];
}

// ---------------------------------------------------------------------------
// 1) Projection GEMM:  out[R,1024] = A @ W + bias, W given pre-transposed bf16.
//    A_MODE:   0 = A row-major [R,1024]      1 = gather from X[B,H,L,D]
//    OUT_MODE: 0 = row-major [R,1024]        1 = scatter to [B,H,L,D]
//              2 = scatter transposed to [B,H,D,L]
// ---------------------------------------------------------------------------
template <int A_MODE, int OUT_MODE>
__global__ __launch_bounds__(256) void proj_gemm_bf16(
    const float* __restrict__ A, const bf16* __restrict__ Wt,
    const float* __restrict__ bias, float* __restrict__ out) {
  __shared__ bf16 sA[128][40];   // [row][k], 32 k (+8 pad)
  __shared__ bf16 sW[128][40];   // [col][k]

  const int tid = threadIdx.x, lane = tid & 31, wave = tid >> 5;
  const int wm = wave >> 1, wn = wave & 1;
  const int rowBase = blockIdx.y * 128;
  const int colBase = blockIdx.x * 128;

  v8f acc[2][4];
#pragma unroll
  for (int a = 0; a < 2; ++a)
#pragma unroll
    for (int b = 0; b < 4; ++b) zero_v8(acc[a][b]);

  for (int k0 = 0; k0 < DM; k0 += 32) {
#pragma unroll
    for (int j = 0; j < 4; ++j) {               // A tile: 128x32 (float4 loads)
      int idx = tid + j * 256;                  // 1024 float4 slots
      int r = idx >> 3, c4 = (idx & 7) * 4;
      int gr = rowBase + r, gc = k0 + c4;
      float4 v;
      if (A_MODE == 0) {
        v = *(const float4*)&A[(size_t)gr * DM + gc];
      } else {                                   // gather from X[B,H,L,D]
        int b_ = gr >> 12, l = gr & 4095;
        int h = gc >> 6, d = gc & 63;
        v = *(const float4*)&A[(((size_t)(b_ * Hc + h) * Lc) + l) * Dc + d];
      }
      cvt_store4(&sA[r][c4], v);
    }
#pragma unroll
    for (int j = 0; j < 2; ++j) {               // W tile: 128 cols x 32 k (b128)
      int idx = tid + j * 256;                  // 512 slots of 8 bf16
      int r = idx >> 2, c8 = (idx & 3) * 8;
      *(uint4*)&sW[r][c8] =
          *(const uint4*)&Wt[(size_t)(colBase + r) * DM + k0 + c8];
    }
    if (k0 + 32 < DM) {                          // prefetch next W chunk
      __builtin_prefetch(&Wt[(size_t)(colBase + (tid >> 1)) * DM + k0 + 32], 0, 3);
    }
    __syncthreads();

    v16bf bfr[4];
#pragma unroll
    for (int tn = 0; tn < 4; ++tn)
      bfr[tn] = lds_frag16(frag_ptr(&sW[0][0], 40, wn * 64 + tn * 16, 0, lane));
#pragma unroll
    for (int tm = 0; tm < 2; ++tm) {
      v16bf af = lds_frag16(frag_ptr(&sA[0][0], 40, wm * 32 + tm * 16, 0, lane));
#pragma unroll
      for (int tn = 0; tn < 4; ++tn)
        acc[tm][tn] = __builtin_amdgcn_wmma_f32_16x16x32_bf16(
            false, af, false, bfr[tn], (short)0, acc[tm][tn], false, false);
    }
    __syncthreads();
  }

#pragma unroll
  for (int tm = 0; tm < 2; ++tm)
#pragma unroll
    for (int tn = 0; tn < 4; ++tn)
#pragma unroll
      for (int r = 0; r < 8; ++r) {
        int row = rowBase + wm * 32 + tm * 16 + r + ((lane >> 4) << 3);
        int col = colBase + wn * 64 + tn * 16 + (lane & 15);
        float v = acc[tm][tn][r] + bias[col];
        if (OUT_MODE == 0) {
          out[(size_t)row * DM + col] = v;
        } else {
          int b_ = row >> 12, l = row & 4095;
          int h = col >> 6, d = col & 63;
          if (OUT_MODE == 1)
            out[(((size_t)(b_ * Hc + h) * Lc) + l) * Dc + d] = v;
          else  // OUT_MODE == 2: [B,H,D,L]
            out[(((size_t)(b_ * Hc + h) * Dc) + d) * Lc + l] = v;
        }
      }
}

// ---------------------------------------------------------------------------
// 2) Landmark pooling: out[bh][m][d] = mean_{j<32} in[bh][m*32+j][d]
// ---------------------------------------------------------------------------
__global__ __launch_bounds__(256) void pool_mean(const float* __restrict__ in,
                                                 float* __restrict__ out) {
  int idx = blockIdx.x * 256 + threadIdx.x;
  int d = idx & 63;
  int m = (idx >> 6) & 127;
  int bh = idx >> 13;
  const float* p = in + ((size_t)bh * Lc + (size_t)m * 32) * Dc + d;
  float s = 0.0f;
#pragma unroll
  for (int j = 0; j < 32; ++j) s += p[(size_t)j * Dc];
  out[idx] = s * (1.0f / 32.0f);
}

// ---------------------------------------------------------------------------
// 3) Batched f32 matmul (one WG per bh): C[128,N] = alpha * A[128,K] @ op(B)
//    V_WMMA_F32_16X16X4_F32.  TRB: B stored [N][K].  OTR: write C transposed.
// ---------------------------------------------------------------------------
template <int N, int K, bool TRB, bool OTR>
__global__ __launch_bounds__(256) void batched_mm_f32(
    float* __restrict__ C, const float* __restrict__ A,
    const float* __restrict__ Bm, float alpha) {
  constexpr int WN = N / 32;
  __shared__ float sA[128][20];   // 16 k (+4 pad -> 80B rows, 16B aligned)
  __shared__ float sB[16][N + 1];

  const int bh = blockIdx.x;
  const float* Ab = A + (size_t)bh * 128 * K;
  const float* Bp = Bm + (size_t)bh * K * N;
  float* Cb = C + (size_t)bh * 128 * N;

  const int tid = threadIdx.x, lane = tid & 31, wave = tid >> 5;
  const int wm = wave >> 1, wn = wave & 1;

  v8f acc[2][WN];
#pragma unroll
  for (int a = 0; a < 2; ++a)
#pragma unroll
    for (int b = 0; b < WN; ++b) zero_v8(acc[a][b]);

  for (int k0 = 0; k0 < K; k0 += 16) {
#pragma unroll
    for (int j = 0; j < 2; ++j) {               // A chunk 128x16 via float4
      int idx = tid + j * 256;                  // 512 float4 slots
      int r = idx >> 2, c4 = (idx & 3) * 4;
      *(float4*)&sA[r][c4] = *(const float4*)&Ab[(size_t)r * K + k0 + c4];
    }
    constexpr int BL = (16 * N) / 256;
#pragma unroll
    for (int j = 0; j < BL; ++j) {              // B chunk 16xN
      int idx = tid + j * 256;
      int r = idx / N, c = idx % N;
      sB[r][c] = TRB ? Bp[(size_t)c * K + k0 + r] : Bp[(size_t)(k0 + r) * N + c];
    }
    __syncthreads();

    const int ks = (lane >> 4) * 2;
#pragma unroll
    for (int kk = 0; kk < 16; kk += 4) {
      v2f bfr[WN];
#pragma unroll
      for (int tn = 0; tn < WN; ++tn) {
        int n = wn * (N / 2) + tn * 16 + (lane & 15);
        bfr[tn][0] = sB[kk + ks][n];
        bfr[tn][1] = sB[kk + ks + 1][n];
      }
#pragma unroll
      for (int tm = 0; tm < 2; ++tm) {
        int r = wm * 32 + tm * 16 + (lane & 15);
        v2f af;
        af[0] = sA[r][kk + ks];
        af[1] = sA[r][kk + ks + 1];
#pragma unroll
        for (int tn = 0; tn < WN; ++tn)
          acc[tm][tn] = __builtin_amdgcn_wmma_f32_16x16x4_f32(
              false, af, false, bfr[tn], (short)0, acc[tm][tn], false, false);
      }
    }
    __syncthreads();
  }

#pragma unroll
  for (int tm = 0; tm < 2; ++tm)
#pragma unroll
    for (int tn = 0; tn < WN; ++tn)
#pragma unroll
      for (int r = 0; r < 8; ++r) {
        int row = wm * 32 + tm * 16 + r + ((lane >> 4) << 3);
        int col = wn * (N / 2) + tn * 16 + (lane & 15);
        if (OTR)
          Cb[(size_t)col * 128 + row] = alpha * acc[tm][tn][r];
        else
          Cb[(size_t)row * N + col] = alpha * acc[tm][tn][r];
      }
}

// out = c*I - in,  layout [BH][128][128]
__global__ __launch_bounds__(256) void diag_sub(float* __restrict__ out,
                                                const float* __restrict__ in,
                                                float c) {
  int idx = blockIdx.x * 256 + threadIdx.x;
  int r = (idx >> 7) & 127, col = idx & 127;
  out[idx] = ((r == col) ? c : 0.0f) - in[idx];
}

// Row softmax over 128 columns; one wave per row.
__global__ __launch_bounds__(256) void softmax128(float* __restrict__ out,
                                                  const float* __restrict__ in) {
  int row = blockIdx.x * 8 + (threadIdx.x >> 5);
  int lane = threadIdx.x & 31;
  const float* r = in + (size_t)row * 128;
  float v[4];
  float mx = -3.0e38f;
#pragma unroll
  for (int j = 0; j < 4; ++j) {
    v[j] = r[lane + 32 * j];
    mx = fmaxf(mx, v[j]);
  }
#pragma unroll
  for (int off = 16; off; off >>= 1) mx = fmaxf(mx, __shfl_xor(mx, off, 32));
  float s = 0.0f;
#pragma unroll
  for (int j = 0; j < 4; ++j) {
    v[j] = __expf(v[j] - mx);
    s += v[j];
  }
#pragma unroll
  for (int off = 16; off; off >>= 1) s += __shfl_xor(s, off, 32);
  float inv = 1.0f / s;
#pragma unroll
  for (int j = 0; j < 4; ++j) out[(size_t)row * 128 + lane + 32 * j] = v[j] * inv;
}

// Newton-Schulz init: V = K^T / max_col(colsum(K)); 128 threads per bh
__global__ __launch_bounds__(128) void ns_init(float* __restrict__ V,
                                               const float* __restrict__ Km) {
  __shared__ float red[128];
  int bh = blockIdx.x, t = threadIdx.x;
  const float* K = Km + (size_t)bh * 16384;
  float* Vb = V + (size_t)bh * 16384;
  float s = 0.0f;
  for (int r = 0; r < 128; ++r) s += K[r * 128 + t];
  red[t] = s;
  __syncthreads();
  for (int off = 64; off; off >>= 1) {
    if (t < off) red[t] = fmaxf(red[t], red[t + off]);
    __syncthreads();
  }
  float scale = 1.0f / red[0];
  for (int j = 0; j < 128; ++j) Vb[t * 128 + j] = scale * K[j * 128 + t];
}

// ---------------------------------------------------------------------------
// 4) Fused kernel_3 stage: T[bh][128][64] = softmax_rows(Ql @ k^T) @ v
//    k in [B,H,L,D]; v pre-transposed [B,H,D,L].  exp-softmax (no max sub).
// ---------------------------------------------------------------------------
__global__ __launch_bounds__(256) void attn_k3v(const float* __restrict__ Ql,
                                                const float* __restrict__ kmat,
                                                const float* __restrict__ vt,
                                                float* __restrict__ T) {
  __shared__ bf16 sKV[9216];       // k tile [128][72] / v tile [64][136]
  __shared__ bf16 sP[128][136];
  __shared__ float rowsum[128];

  const int bh = blockIdx.x;
  const int tid = threadIdx.x, lane = tid & 31, wave = tid >> 5;
  const int wm = wave >> 1, wn = wave & 1;
  const int hi8 = (lane >> 4) << 3;

  // Ql A-fragments pinned in registers (rows wm*32..+32, K = 64 in 2 chunks)
  const float* Qb = Ql + (size_t)bh * Mc * Dc;
  v16bf qfrag[2][2];
#pragma unroll
  for (int tm = 0; tm < 2; ++tm)
#pragma unroll
    for (int kc = 0; kc < 2; ++kc) {
      int row = wm * 32 + tm * 16 + (lane & 15);
      qfrag[tm][kc] = global_frag_f32(Qb + (size_t)row * Dc + kc * 32 + hi8);
    }

  if (tid < 128) rowsum[tid] = 0.0f;

  v8f tacc[2][2];
#pragma unroll
  for (int a = 0; a < 2; ++a)
#pragma unroll
    for (int b = 0; b < 2; ++b) zero_v8(tacc[a][b]);

  const float* kg = kmat + (size_t)bh * Lc * Dc;
  const float* vg = vt + (size_t)bh * Dc * Lc;

  for (int l0 = 0; l0 < Lc; l0 += 128) {
    __syncthreads();
#pragma unroll
    for (int j = 0; j < 8; ++j) {               // k tile [128][64] -> [r][c]
      int idx = tid + j * 256;                  // 2048 float4 slots
      int r = idx >> 4, c4 = (idx & 15) * 4;
      float4 v = *(const float4*)&kg[(size_t)(l0 + r) * Dc + c4];
      cvt_store4(&sKV[r * 72 + c4], v);
    }
    __syncthreads();

    // S = Ql @ k^T : wave rows wm*32 (2 tiles), cols wn*64 (4 tiles)
    v8f sacc[2][4];
#pragma unroll
    for (int a = 0; a < 2; ++a)
#pragma unroll
      for (int b = 0; b < 4; ++b) zero_v8(sacc[a][b]);
#pragma unroll
    for (int kc = 0; kc < 2; ++kc)
#pragma unroll
      for (int tn = 0; tn < 4; ++tn) {
        v16bf bfr =
            lds_frag16(frag_ptr(sKV, 72, wn * 64 + tn * 16, kc * 32, lane));
#pragma unroll
        for (int tm = 0; tm < 2; ++tm)
          sacc[tm][tn] = __builtin_amdgcn_wmma_f32_16x16x32_bf16(
              false, qfrag[tm][kc], false, bfr, (short)0, sacc[tm][tn], false,
              false);
      }

    // P = exp(S) into LDS (bf16)
#pragma unroll
    for (int tm = 0; tm < 2; ++tm)
#pragma unroll
      for (int tn = 0; tn < 4; ++tn)
#pragma unroll
        for (int r = 0; r < 8; ++r) {
          int mrow = wm * 32 + tm * 16 + r + ((lane >> 4) << 3);
          int col = wn * 64 + tn * 16 + (lane & 15);
          sP[mrow][col] = (bf16)__expf(sacc[tm][tn][r]);
        }
    __syncthreads();

    if (tid < 128) {                             // row sums, 16B LDS reads
      const uint4* pr = (const uint4*)&sP[tid][0];
      float s = 0.0f;
#pragma unroll
      for (int j = 0; j < 16; ++j) {
        uint4 u = pr[j];
        s += bf16pair_sum(u.x) + bf16pair_sum(u.y) + bf16pair_sum(u.z) +
             bf16pair_sum(u.w);
      }
      rowsum[tid] += s;
    }
#pragma unroll
    for (int j = 0; j < 8; ++j) {               // v tile [64][128] from [B,H,D,L]
      int idx = tid + j * 256;                  // 2048 float4 slots
      int d = idx >> 5, l4 = (idx & 31) * 4;
      float4 v = *(const float4*)&vg[(size_t)d * Lc + l0 + l4];
      cvt_store4(&sKV[d * 136 + l4], v);
    }
    __syncthreads();

    // T += P @ v : wave rows wm*32 (2 tiles), cols wn*32 (2 tiles), K=128
#pragma unroll
    for (int kc = 0; kc < 4; ++kc) {
      v16bf pf[2];
#pragma unroll
      for (int tm = 0; tm < 2; ++tm)
        pf[tm] =
            lds_frag16(frag_ptr(&sP[0][0], 136, wm * 32 + tm * 16, kc * 32, lane));
#pragma unroll
      for (int tn = 0; tn < 2; ++tn) {
        v16bf vf =
            lds_frag16(frag_ptr(sKV, 136, wn * 32 + tn * 16, kc * 32, lane));
#pragma unroll
        for (int tm = 0; tm < 2; ++tm)
          tacc[tm][tn] = __builtin_amdgcn_wmma_f32_16x16x32_bf16(
              false, pf[tm], false, vf, (short)0, tacc[tm][tn], false, false);
      }
    }
  }

  __syncthreads();
  if (tid < 128) rowsum[tid] = 1.0f / rowsum[tid];
  __syncthreads();

  float* Tb = T + (size_t)bh * Mc * Dc;
#pragma unroll
  for (int tm = 0; tm < 2; ++tm)
#pragma unroll
    for (int tn = 0; tn < 2; ++tn)
#pragma unroll
      for (int r = 0; r < 8; ++r) {
        int row = wm * 32 + tm * 16 + r + ((lane >> 4) << 3);
        int col = wn * 32 + tn * 16 + (lane & 15);
        Tb[(size_t)row * Dc + col] = tacc[tm][tn][r] * rowsum[row];
      }
}

// ---------------------------------------------------------------------------
// 5) Fused kernel_1 stage: X_tile = softmax_rows(q_tile @ Kl^T) @ Z
//    Z pre-transposed [bh][64][128]; X written in [B,H,L,D].
// ---------------------------------------------------------------------------
__global__ __launch_bounds__(256) void attn_out(const float* __restrict__ q,
                                                const float* __restrict__ Kl,
                                                const float* __restrict__ Zt,
                                                float* __restrict__ X) {
  __shared__ bf16 sKZ[9216];       // Kl tile [128][72] / Z tile [64][136]
  __shared__ bf16 sP[128][136];
  __shared__ float rowsum[128];

  const int bh = blockIdx.y, rt = blockIdx.x;
  const int tid = threadIdx.x, lane = tid & 31, wave = tid >> 5;
  const int wm = wave >> 1, wn = wave & 1;
  const int hi8 = (lane >> 4) << 3;

  const float* qb = q + ((size_t)bh * Lc + (size_t)rt * 128) * Dc;
  v16bf qfrag[2][2];
#pragma unroll
  for (int tm = 0; tm < 2; ++tm)
#pragma unroll
    for (int kc = 0; kc < 2; ++kc) {
      int row = wm * 32 + tm * 16 + (lane & 15);
      qfrag[tm][kc] = global_frag_f32(qb + (size_t)row * Dc + kc * 32 + hi8);
    }

  const float* Klb = Kl + (size_t)bh * Mc * Dc;
#pragma unroll
  for (int j = 0; j < 8; ++j) {                 // Kl tile [128][64]
    int idx = tid + j * 256;
    int r = idx >> 4, c4 = (idx & 15) * 4;
    float4 v = *(const float4*)&Klb[(size_t)r * Dc + c4];
    cvt_store4(&sKZ[r * 72 + c4], v);
  }
  __syncthreads();

  v8f sacc[2][4];
#pragma unroll
  for (int a = 0; a < 2; ++a)
#pragma unroll
    for (int b = 0; b < 4; ++b) zero_v8(sacc[a][b]);
#pragma unroll
  for (int kc = 0; kc < 2; ++kc)
#pragma unroll
    for (int tn = 0; tn < 4; ++tn) {
      v16bf bfr =
          lds_frag16(frag_ptr(sKZ, 72, wn * 64 + tn * 16, kc * 32, lane));
#pragma unroll
      for (int tm = 0; tm < 2; ++tm)
        sacc[tm][tn] = __builtin_amdgcn_wmma_f32_16x16x32_bf16(
            false, qfrag[tm][kc], false, bfr, (short)0, sacc[tm][tn], false,
            false);
    }

#pragma unroll
  for (int tm = 0; tm < 2; ++tm)
#pragma unroll
    for (int tn = 0; tn < 4; ++tn)
#pragma unroll
      for (int r = 0; r < 8; ++r) {
        int mrow = wm * 32 + tm * 16 + r + ((lane >> 4) << 3);
        int col = wn * 64 + tn * 16 + (lane & 15);
        sP[mrow][col] = (bf16)__expf(sacc[tm][tn][r]);
      }
  __syncthreads();

  if (tid < 128) {
    const uint4* pr = (const uint4*)&sP[tid][0];
    float s = 0.0f;
#pragma unroll
    for (int j = 0; j < 16; ++j) {
      uint4 u = pr[j];
      s += bf16pair_sum(u.x) + bf16pair_sum(u.y) + bf16pair_sum(u.z) +
           bf16pair_sum(u.w);
    }
    rowsum[tid] = 1.0f / s;
  }
  const float* Zb = Zt + (size_t)bh * Dc * Mc;   // [64][128]
#pragma unroll
  for (int j = 0; j < 8; ++j) {
    int idx = tid + j * 256;
    int d = idx >> 5, m4 = (idx & 31) * 4;
    float4 v = *(const float4*)&Zb[(size_t)d * Mc + m4];
    cvt_store4(&sKZ[d * 136 + m4], v);
  }
  __syncthreads();

  v8f tacc[2][2];
#pragma unroll
  for (int a = 0; a < 2; ++a)
#pragma unroll
    for (int b = 0; b < 2; ++b) zero_v8(tacc[a][b]);
#pragma unroll
  for (int kc = 0; kc < 4; ++kc) {
    v16bf pf[2];
#pragma unroll
    for (int tm = 0; tm < 2; ++tm)
      pf[tm] =
          lds_frag16(frag_ptr(&sP[0][0], 136, wm * 32 + tm * 16, kc * 32, lane));
#pragma unroll
    for (int tn = 0; tn < 2; ++tn) {
      v16bf vf =
          lds_frag16(frag_ptr(sKZ, 136, wn * 32 + tn * 16, kc * 32, lane));
#pragma unroll
      for (int tm = 0; tm < 2; ++tm)
        tacc[tm][tn] = __builtin_amdgcn_wmma_f32_16x16x32_bf16(
            false, pf[tm], false, vf, (short)0, tacc[tm][tn], false, false);
    }
  }

  float* Xb = X + ((size_t)bh * Lc + (size_t)rt * 128) * Dc;
#pragma unroll
  for (int tm = 0; tm < 2; ++tm)
#pragma unroll
    for (int tn = 0; tn < 2; ++tn)
#pragma unroll
      for (int r = 0; r < 8; ++r) {
        int row = wm * 32 + tm * 16 + r + ((lane >> 4) << 3);
        int col = wn * 32 + tn * 16 + (lane & 15);
        Xb[(size_t)row * Dc + col] = tacc[tm][tn][r] * rowsum[row];
      }
}

// ---------------------------------------------------------------------------
// Host orchestration
// ---------------------------------------------------------------------------
extern "C" void kernel_launch(void* const* d_in, const int* in_sizes, int n_in,
                              void* d_out, int out_size, void* d_ws,
                              size_t ws_size, hipStream_t stream) {
  const float* queries = (const float*)d_in[0];
  const float* keys    = (const float*)d_in[1];
  const float* values  = (const float*)d_in[2];
  const float* Wq = (const float*)d_in[3];
  const float* bq = (const float*)d_in[4];
  const float* Wk = (const float*)d_in[5];
  const float* bk = (const float*)d_in[6];
  const float* Wv = (const float*)d_in[7];
  const float* bv = (const float*)d_in[8];
  const float* Wo = (const float*)d_in[9];
  const float* bo = (const float*)d_in[10];
  float* out = (float*)d_out;

  char* wsb = (char*)d_ws;
  size_t off = 0;
  auto take = [&](size_t bytes) -> char* {
    char* p = wsb + off;
    off += (bytes + 255) & ~(size_t)255;
    return p;
  };
  const size_t big = (size_t)Rc * DM * sizeof(float);       // 67 MB
  const size_t lmk = (size_t)BHc * Mc * Dc * sizeof(float); // 2 MB
  const size_t mat = (size_t)BHc * Mc * Mc * sizeof(float); // 4 MB
  const size_t wtb = (size_t)DM * DM * sizeof(bf16);        // 2 MB

  float* q   = (float*)take(big);
  float* k   = (float*)take(big);
  float* vt  = (float*)take(big);   // v transposed [B,H,D,L]
  float* X   = (float*)take(big);
  float* Ql  = (float*)take(lmk);
  float* Klm = (float*)take(lmk);
  float* S2  = (float*)take(mat);
  float* Kmat = (float*)take(mat);
  float* Vb  = (float*)take(mat);
  float* KV  = (float*)take(mat);
  float* E1  = (float*)take(mat);
  float* E2  = (float*)take(mat);
  float* T   = (float*)take(lmk);
  float* Zt  = (float*)take(lmk);
  bf16* WqT = (bf16*)take(wtb);
  bf16* WkT = (bf16*)take(wtb);
  bf16* WvT = (bf16*)take(wtb);
  bf16* WoT = (bf16*)take(wtb);

  // 0) transpose + bf16-convert weights
  transpose_w<<<1024, 256, 0, stream>>>(Wq, WqT);
  transpose_w<<<1024, 256, 0, stream>>>(Wk, WkT);
  transpose_w<<<1024, 256, 0, stream>>>(Wv, WvT);
  transpose_w<<<1024, 256, 0, stream>>>(Wo, WoT);

  dim3 gGemm(DM / 128, Rc / 128);   // 8 x 128

  // 1) projections
  proj_gemm_bf16<0, 1><<<gGemm, 256, 0, stream>>>(queries, WqT, bq, q);
  proj_gemm_bf16<0, 1><<<gGemm, 256, 0, stream>>>(keys, WkT, bk, k);
  proj_gemm_bf16<0, 2><<<gGemm, 256, 0, stream>>>(values, WvT, bv, vt);

  // 2) landmark pooling
  pool_mean<<<(BHc * Mc * Dc) / 256, 256, 0, stream>>>(q, Ql);
  pool_mean<<<(BHc * Mc * Dc) / 256, 256, 0, stream>>>(k, Klm);

  // 3) kernel_2 = softmax(Ql @ Kl^T), Newton-Schulz inverse
  batched_mm_f32<128, 64, true, false><<<BHc, 256, 0, stream>>>(S2, Ql, Klm, 1.0f);
  softmax128<<<(BHc * Mc) / 8, 256, 0, stream>>>(Kmat, S2);
  ns_init<<<BHc, 128, 0, stream>>>(Vb, Kmat);

  float* Vc = Vb;
  float* Sp = E2;
  for (int it = 0; it < 6; ++it) {
    batched_mm_f32<128, 128, false, false><<<BHc, 256, 0, stream>>>(KV, Kmat, Vc, 1.0f);
    diag_sub<<<(BHc * Mc * Mc) / 256, 256, 0, stream>>>(E1, KV, 7.0f);
    batched_mm_f32<128, 128, false, false><<<BHc, 256, 0, stream>>>(Sp, KV, E1, 1.0f);
    diag_sub<<<(BHc * Mc * Mc) / 256, 256, 0, stream>>>(E1, Sp, 15.0f);
    batched_mm_f32<128, 128, false, false><<<BHc, 256, 0, stream>>>(Sp, KV, E1, 1.0f);
    diag_sub<<<(BHc * Mc * Mc) / 256, 256, 0, stream>>>(E1, Sp, 13.0f);
    batched_mm_f32<128, 128, false, false><<<BHc, 256, 0, stream>>>(Sp, Vc, E1, 0.25f);
    float* t = Vc; Vc = Sp; Sp = t;
  }

  // 4) T = softmax(Ql @ k^T) @ v
  attn_k3v<<<BHc, 256, 0, stream>>>(Ql, k, vt, T);

  // 5) Zt = (Vc @ T)^T  -> [bh][64][128]
  batched_mm_f32<64, 128, false, true><<<BHc, 256, 0, stream>>>(Zt, Vc, T, 1.0f);

  // 6) X = softmax(q @ Kl^T) @ Z   (X in [B,H,L,D])
  attn_out<<<dim3(Lc / 128, BHc), 256, 0, stream>>>(q, Klm, Zt, X);

  // 7) out = X @ Wo + bo
  proj_gemm_bf16<1, 0><<<gGemm, 256, 0, stream>>>(X, WoT, bo, out);
}
